// RNNPolicy_17136919511666
// MI455X (gfx1250) — compile-verified, matrix operand order checked by reading, not run on hardware
//
#include <hip/hip_runtime.h>
#include <hip/hip_bf16.h>

typedef __attribute__((ext_vector_type(16))) __bf16 v16bf;
typedef __attribute__((ext_vector_type(8)))  float  v8f;

#define B_   32
#define T_   2048
#define OBS_ 64
#define FC_  256
#define ACT_ 100
#define INP_ 384   // 356 padded to multiple of 32
#define H_   512
#define G4_  2048  // 4*H
#define KTX  12    // INP_/32
#define KTH  16    // H_/32
#define NT4  128   // G4_/16
#define NTP  7     // proj N: 100 padded to 112 = 7*16
#define MT_  4096  // (B_*T_)/16 row tiles
#define MP_  2048  // M-tile pairs
#define NWG_LSTM 16

#if defined(__gfx1250__)
#define USE_ASYNC_LDS 1
#else
#define USE_ASYNC_LDS 0
#endif

#define WMMA_BF16(a, b, c) \
  __builtin_amdgcn_wmma_f32_16x16x32_bf16(false, (a), false, (b), (short)0, (c), false, false)

// ---------------------------------------------------------------------------
// ISA 16-bit A-fragment placement (CDNA5 ISA §7.12.2):
// lanes 0-15 hold K in {0..7} (slots 0-7) and {16..23} (slots 8-15);
// lanes 16-31 hold K in {8..15} and {24..31}. Row m -> lane base m.
// Fragments stored as 32 lanes x 16 contiguous halfs (512 halfs / tile).
// ---------------------------------------------------------------------------
__device__ __forceinline__ void amap(int m, int k, int& lane, int& slot) {
  if (k < 8)       { lane = m;      slot = k;      }
  else if (k < 16) { lane = m + 16; slot = k - 8;  }
  else if (k < 24) { lane = m;      slot = k - 8;  }
  else             { lane = m + 16; slot = k - 16; }
}

__device__ __forceinline__ float sigm(float x) { return 1.f / (1.f + __expf(-x)); }
__device__ __forceinline__ float tanh_(float x) {
  float e = __expf(-2.f * x);
  return (1.f - e) / (1.f + e);
}

// Sense-reversing device-scope barrier across NWG_LSTM workgroups.
__device__ __forceinline__ void gbar(int* bar, int nwg) {
  __syncthreads();
  if (threadIdx.x == 0) {
    int gen = __hip_atomic_load(&bar[1], __ATOMIC_ACQUIRE, __HIP_MEMORY_SCOPE_AGENT);
    if (__hip_atomic_fetch_add(&bar[0], 1, __ATOMIC_ACQ_REL, __HIP_MEMORY_SCOPE_AGENT) == nwg - 1) {
      __hip_atomic_store(&bar[0], 0, __ATOMIC_RELAXED, __HIP_MEMORY_SCOPE_AGENT);
      __hip_atomic_fetch_add(&bar[1], 1, __ATOMIC_RELEASE, __HIP_MEMORY_SCOPE_AGENT);
    } else {
      while (__hip_atomic_load(&bar[1], __ATOMIC_ACQUIRE, __HIP_MEMORY_SCOPE_AGENT) == gen)
        __builtin_amdgcn_s_sleep(2);
    }
  }
  __syncthreads();
}

__global__ void k_init(int* bar) {
  if (threadIdx.x == 0) { bar[0] = 0; bar[1] = 0; }
}

// Convert a row-major fp32 [srcK, srcN] weight matrix into bf16 B-fragment
// tiles (mirrored A layout, col n = lane&15), zero-padding K and N.
__global__ void k_cvt_bfrag(const float* __restrict__ src, int srcK, int srcN, int ldN,
                            int KT, __bf16* __restrict__ dst, int total) {
  int id = blockIdx.x * blockDim.x + threadIdx.x;
  if (id >= total) return;
  int r = id & 511, frag = id >> 9;
  int lane = r >> 4, slot = r & 15;
  int kt = frag % KT, nt = frag / KT;
  int n = lane & 15;
  int k = (lane < 16) ? ((slot < 8) ? slot : slot + 8)
                      : ((slot < 8) ? slot + 8 : slot + 16);
  int kg = kt * 32 + k, ng = nt * 16 + n;
  float v = (kg < srcK && ng < srcN) ? src[(size_t)kg * ldN + ng] : 0.f;
  dst[id] = (__bf16)v;
}

// Fused feature build: x = concat(relu(obs@W_obs+b), relu(act*W_act+b), 0-pad)
// written directly as bf16 A-fragments, row order r = t*32 + b.
// obs tile staged into LDS with GLOBAL_LOAD_ASYNC_TO_LDS_B128 (inline asm,
// the portable path per the bridge doc), tracked with ASYNCcnt.
__global__ void k_prep(const float* __restrict__ obs, const float* __restrict__ act,
                       const float* __restrict__ Wobs, const float* __restrict__ bobs,
                       const float* __restrict__ Wact, const float* __restrict__ bact,
                       __bf16* __restrict__ xf) {
  __shared__ float sobs[16][OBS_];
  __shared__ float sact[16];
  int mt = blockIdx.x, tid = threadIdx.x;
#if USE_ASYNC_LDS
  {
    // 16 rows x 256B each; one b128 async copy per thread (256 threads x 16B).
    int m = tid >> 4, c = tid & 15;           // row in tile, 16B chunk
    int r = mt * 16 + m;
    const float* g = obs + ((size_t)(r & 31) * T_ + (r >> 5)) * OBS_ + c * 4;
    unsigned lofs = (unsigned)(size_t)(__attribute__((address_space(3))) void*)&sobs[m][c * 4];
    asm volatile("global_load_async_to_lds_b128 %0, %1, off"
                 :
                 : "v"(lofs), "v"(g)
                 : "memory");
  }
#else
  for (int i = tid; i < 16 * OBS_; i += 256) {
    int m = i >> 6, d = i & 63;
    int r = mt * 16 + m;
    sobs[m][d] = obs[((size_t)(r & 31) * T_ + (r >> 5)) * OBS_ + d];
  }
#endif
  if (tid < 16) {
    int r = mt * 16 + tid;
    sact[tid] = act[(size_t)(r & 31) * T_ + (r >> 5)];
  }
#if USE_ASYNC_LDS
  asm volatile("s_wait_asynccnt 0x0" ::: "memory");
#endif
  __syncthreads();
  for (int j = tid; j < INP_; j += 256) {
    int kt = j >> 5, kl = j & 31;
    if (j < FC_) {
      float wc[OBS_];
      for (int d = 0; d < OBS_; ++d) wc[d] = Wobs[(size_t)d * FC_ + j];
      float bj = bobs[j];
      for (int m = 0; m < 16; ++m) {
        float v = bj;
        for (int d = 0; d < OBS_; ++d) v += sobs[m][d] * wc[d];
        v = v > 0.f ? v : 0.f;
        int lane, slot; amap(m, kl, lane, slot);
        xf[((size_t)mt * KTX + kt) * 512 + lane * 16 + slot] = (__bf16)v;
      }
    } else {
      int jj = j - FC_;
      float w = (jj < ACT_) ? Wact[jj] : 0.f;
      float ba = (jj < ACT_) ? bact[jj] : 0.f;
      for (int m = 0; m < 16; ++m) {
        float v = 0.f;
        if (jj < ACT_) { v = sact[m] * w + ba; v = v > 0.f ? v : 0.f; }
        int lane, slot; amap(m, kl, lane, slot);
        xf[((size_t)mt * KTX + kt) * 512 + lane * 16 + slot] = (__bf16)v;
      }
    }
  }
}

// Z[t*32+b, 0:2048] = x @ W_lstm + b_lstm. Register-blocked 2 M-tiles x
// 4 N-tiles per wave (8 WMMAs per k-step on 6 fragment loads). launch_bounds
// unlocks the full VGPR file so the 64 accumulator VGPRs never spill.
// Z is streamed with non-temporal stores (512 MB, no reuse before eviction).
__global__ void __launch_bounds__(256, 1)
k_gemm_z(const __bf16* __restrict__ xf, const __bf16* __restrict__ wlf,
         const float* __restrict__ blstm, float* __restrict__ Z) {
  int mp = blockIdx.x, ng = blockIdx.y;
  int wid = threadIdx.x >> 5, lane = threadIdx.x & 31;
  int nt0 = ng * 32 + wid * 4;
  const __bf16* a0p = xf + ((size_t)(2 * mp) * KTX) * 512 + lane * 16;
  const __bf16* a1p = a0p + (size_t)KTX * 512;
  const __bf16* bp  = wlf + ((size_t)nt0 * KTX) * 512 + lane * 16;
  v8f acc[2][4] = {};
#pragma unroll
  for (int kt = 0; kt < KTX; ++kt) {
    v16bf a0 = *(const v16bf*)(a0p + kt * 512);
    v16bf a1 = *(const v16bf*)(a1p + kt * 512);
#pragma unroll
    for (int j = 0; j < 4; ++j) {
      v16bf b = *(const v16bf*)(bp + ((size_t)j * KTX + kt) * 512);
      acc[0][j] = WMMA_BF16(a0, b, acc[0][j]);
      acc[1][j] = WMMA_BF16(a1, b, acc[1][j]);
    }
  }
  int mrow = (lane >> 4) * 8;
#pragma unroll
  for (int j = 0; j < 4; ++j) {
    int n = (nt0 + j) * 16 + (lane & 15);
    float bias = blstm[n];
#pragma unroll
    for (int h = 0; h < 2; ++h) {
#pragma unroll
      for (int g = 0; g < 8; ++g) {
        int row = (2 * mp + h) * 16 + mrow + g;
        __builtin_nontemporal_store(acc[h][j][g] + bias, &Z[(size_t)row * G4_ + n]);
      }
    }
  }
}

// Persistent recurrent kernel: 16 WGs x 8 waves. Wave owns one 16-col slice
// of 4H; U B-fragments live in VGPRs for all 2048 steps. h published in
// WMMA A-fragment layout (seqf) so next step + projection read it directly.
// Next step's Z slab is prefetched (global_prefetch) during the gate phase;
// Z itself is read with non-temporal hints (streamed once).
__global__ void __launch_bounds__(256, 1)
k_lstm(const float* __restrict__ Z, const __bf16* __restrict__ uf,
       __bf16* __restrict__ seqf, float* __restrict__ zbuf,
       int* __restrict__ bar) {
  const int wg = blockIdx.x, tid = threadIdx.x;
  const int wid = tid >> 5, lane = tid & 31;
  const int nt = wg * 8 + wid;
  const int ncol = nt * 16 + (lane & 15);
  const int mrow = (lane >> 4) * 8;
  __shared__ float c_lds[1024];  // cell state for this WG's 32 h-columns x 32 batch

  v16bf ub[KTH];
#pragma unroll
  for (int kt = 0; kt < KTH; ++kt)
    ub[kt] = *(const v16bf*)(uf + ((size_t)nt * KTH + kt) * 512 + lane * 16);

  for (int t = 0; t < T_; ++t) {
    const size_t zb = (size_t)t * B_ * G4_;
    v8f acc0, acc1;
#pragma unroll
    for (int g = 0; g < 8; ++g) {
      acc0[g] = __builtin_nontemporal_load(&Z[zb + (size_t)(mrow + g) * G4_ + ncol]);
      acc1[g] = __builtin_nontemporal_load(&Z[zb + (size_t)(16 + mrow + g) * G4_ + ncol]);
    }
    if (t > 0) {
      const __bf16* s0 = seqf + ((size_t)(2 * t - 2) * KTH) * 512 + lane * 16;
      const __bf16* s1 = s0 + (size_t)KTH * 512;
#pragma unroll
      for (int kt = 0; kt < KTH; ++kt) {
        v16bf a0 = *(const v16bf*)(s0 + kt * 512);
        acc0 = WMMA_BF16(a0, ub[kt], acc0);
        v16bf a1 = *(const v16bf*)(s1 + kt * 512);
        acc1 = WMMA_BF16(a1, ub[kt], acc1);
      }
    }
#pragma unroll
    for (int g = 0; g < 8; ++g) {
      zbuf[(size_t)(mrow + g) * G4_ + ncol] = acc0[g];
      zbuf[(size_t)(16 + mrow + g) * G4_ + ncol] = acc1[g];
    }
    __threadfence();
    gbar(bar, NWG_LSTM);  // all gate pre-activations visible

    // Pull next step's Z slab toward the caches while doing gate math.
    if (t + 1 < T_)
      __builtin_prefetch(&Z[(size_t)(t + 1) * B_ * G4_ + ((size_t)wg * 256 + tid) * 16], 0, 0);

    // Gate math: WG owns h columns [32*wg, 32*wg+32), all 32 batch rows.
#pragma unroll
    for (int e = 0; e < 4; ++e) {
      int idx = tid * 4 + e;
      int bb = idx >> 5;
      int hc = wg * 32 + (idx & 31);
      float zi = zbuf[(size_t)bb * G4_ + hc];
      float zf = zbuf[(size_t)bb * G4_ + 512 + hc];
      float zg = zbuf[(size_t)bb * G4_ + 1024 + hc];
      float zo = zbuf[(size_t)bb * G4_ + 1536 + hc];
      float ig = sigm(zi), fg = sigm(zf), gg = tanh_(zg), og = sigm(zo);
      float cp = (t == 0) ? 0.f : c_lds[idx];
      float cn = fg * cp + ig * gg;
      float h  = og * tanh_(cn);
      c_lds[idx] = cn;
      int kt2 = hc >> 5, kl = hc & 31, lane2, slot;
      amap(bb & 15, kl, lane2, slot);
      seqf[((size_t)(2 * t + (bb >> 4)) * KTH + kt2) * 512 + lane2 * 16 + slot] = (__bf16)h;
    }
    __threadfence();
    gbar(bar, NWG_LSTM);  // h published before next step's WMMA reads it
  }
}

// logits = seq @ W_proj + b_proj. 2 M-tiles per wave so each W_proj B-frag
// load feeds two WMMAs. N padded to 112, writes masked to n<100.
__global__ void __launch_bounds__(224, 1)
k_proj(const __bf16* __restrict__ seqf, const __bf16* __restrict__ wpf,
       const float* __restrict__ bproj, float* __restrict__ logits) {
  int mp = blockIdx.x;
  int nt = threadIdx.x >> 5, lane = threadIdx.x & 31;
  const __bf16* a0p = seqf + ((size_t)(2 * mp) * KTH) * 512 + lane * 16;
  const __bf16* a1p = a0p + (size_t)KTH * 512;
  const __bf16* bp  = wpf + ((size_t)nt * KTH) * 512 + lane * 16;
  v8f acc0 = {}, acc1 = {};
#pragma unroll
  for (int kt = 0; kt < KTH; ++kt) {
    v16bf b = *(const v16bf*)(bp + kt * 512);
    v16bf a0 = *(const v16bf*)(a0p + kt * 512);
    acc0 = WMMA_BF16(a0, b, acc0);
    v16bf a1 = *(const v16bf*)(a1p + kt * 512);
    acc1 = WMMA_BF16(a1, b, acc1);
  }
  int n = nt * 16 + (lane & 15);
  if (n < ACT_) {
    float bias = bproj[n];
    int mrow = (lane >> 4) * 8;
#pragma unroll
    for (int h = 0; h < 2; ++h) {
      for (int g = 0; g < 8; ++g) {
        int r = (2 * mp + h) * 16 + mrow + g;  // r = t*32 + b
        int t = r >> 5, b = r & 31;
        logits[((size_t)b * T_ + t) * ACT_ + n] = ((h == 0) ? acc0[g] : acc1[g]) + bias;
      }
    }
  }
}

__global__ void k_softmax(const float* __restrict__ logits, float* __restrict__ pi) {
  int row = blockIdx.x * blockDim.x + threadIdx.x;
  if (row >= B_ * T_) return;
  const float* l = logits + (size_t)row * ACT_;
  float m = -3.4e38f;
  for (int n = 0; n < ACT_; ++n) m = fmaxf(m, l[n]);
  float s = 0.f;
  for (int n = 0; n < ACT_; ++n) s += __expf(l[n] - m);
  float inv = 1.f / s;
  float* p = pi + (size_t)row * ACT_;
  for (int n = 0; n < ACT_; ++n) p[n] = __expf(l[n] - m) * inv;
}

// argmax over batch axis (axis=0), first-max semantics like jnp.argmax.
__global__ void k_argmax(const float* __restrict__ logits, float* __restrict__ am) {
  int gid = blockIdx.x * blockDim.x + threadIdx.x;
  if (gid >= T_ * ACT_) return;
  int t = gid / ACT_, n = gid % ACT_;
  float best = logits[(size_t)t * ACT_ + n];
  int bi = 0;
  for (int b = 1; b < B_; ++b) {
    float v = logits[((size_t)b * T_ + t) * ACT_ + n];
    if (v > best) { best = v; bi = b; }
  }
  am[gid] = (float)bi;
}

extern "C" void kernel_launch(void* const* d_in, const int* in_sizes, int n_in,
                              void* d_out, int out_size, void* d_ws, size_t ws_size,
                              hipStream_t stream) {
  (void)in_sizes; (void)n_in; (void)out_size; (void)ws_size;
  const float* obs  = (const float*)d_in[0];
  const float* act  = (const float*)d_in[1];
  const float* Wobs = (const float*)d_in[2];
  const float* bobs = (const float*)d_in[3];
  const float* Wact = (const float*)d_in[4];
  const float* bact = (const float*)d_in[5];
  const float* Wl   = (const float*)d_in[6];
  const float* Ul   = (const float*)d_in[7];
  const float* bl   = (const float*)d_in[8];
  const float* Wp   = (const float*)d_in[9];
  const float* bpj  = (const float*)d_in[10];

  char* ws = (char*)d_ws;
  size_t off = 0;
  auto alloc = [&](size_t bytes) -> void* {
    void* p = ws + off;
    off = (off + bytes + 255) & ~(size_t)255;
    return p;
  };
  __bf16* xf   = (__bf16*)alloc((size_t)MT_ * KTX * 512 * 2);   // ~50 MB
  __bf16* wlf  = (__bf16*)alloc((size_t)NT4 * KTX * 512 * 2);   // 1.5 MB
  __bf16* uf   = (__bf16*)alloc((size_t)NT4 * KTH * 512 * 2);   // 2 MB
  __bf16* wpf  = (__bf16*)alloc((size_t)NTP * KTH * 512 * 2);   // 0.1 MB
  float*  Z    = (float*) alloc((size_t)B_ * T_ * G4_ * 4);     // 512 MB
  __bf16* seqf = (__bf16*)alloc((size_t)MT_ * KTH * 512 * 2);   // 64 MB
  float*  zbuf = (float*) alloc((size_t)B_ * G4_ * 4);          // 256 KB
  int*    bar  = (int*)   alloc(64);

  float* out    = (float*)d_out;
  float* pi     = out;
  float* logits = out + (size_t)B_ * T_ * ACT_;
  float* am     = out + 2 * (size_t)B_ * T_ * ACT_;

  k_init<<<1, 32, 0, stream>>>(bar);
  { int tot = NT4 * KTX * 512;
    k_cvt_bfrag<<<(tot + 255) / 256, 256, 0, stream>>>(Wl, 356, G4_, G4_, KTX, wlf, tot); }
  { int tot = NT4 * KTH * 512;
    k_cvt_bfrag<<<(tot + 255) / 256, 256, 0, stream>>>(Ul, H_, G4_, G4_, KTH, uf, tot); }
  { int tot = NTP * KTH * 512;
    k_cvt_bfrag<<<(tot + 255) / 256, 256, 0, stream>>>(Wp, H_, ACT_, ACT_, KTH, wpf, tot); }
  k_prep<<<MT_, 256, 0, stream>>>(obs, act, Wobs, bobs, Wact, bact, xf);
  k_gemm_z<<<dim3(MP_, 4), 256, 0, stream>>>(xf, wlf, bl, Z);
  k_lstm<<<NWG_LSTM, 256, 0, stream>>>(Z, uf, seqf, zbuf, bar);
  k_proj<<<MP_, 224, 0, stream>>>(seqf, wpf, bpj, logits);
  k_softmax<<<(B_ * T_ + 255) / 256, 256, 0, stream>>>(logits, pi);
  k_argmax<<<(T_ * ACT_ + 255) / 256, 256, 0, stream>>>(logits, am);
}
